// LSDDNet_56856777064464
// MI455X (gfx1250) — compile-verified, matrix operand order checked by reading, not run on hardware
//
#include <hip/hip_runtime.h>
#include <hip/hip_bf16.h>

// ---------------------------------------------------------------------------
// Model constants (from reference)
// ---------------------------------------------------------------------------
#define D_MODEL   256
#define D_STATE   16
#define N_LAYERS  4
#define D_CONV    4
#define MLP_HID   128
#define D_INNER   512           // 2*D_MODEL
#define DT_RANK   16
#define BATCH     8
#define SEQ       2048
#define NTOK      (BATCH*SEQ)   // 16384 tokens per encoder
#define DBC_DIM   (DT_RANK + 2*D_STATE)   // 48

typedef __bf16 bf16;
typedef bf16  bf16x16 __attribute__((ext_vector_type(16)));
typedef float f32x8   __attribute__((ext_vector_type(8)));
typedef unsigned int u32x4 __attribute__((ext_vector_type(4)));
typedef int   i32x4   __attribute__((ext_vector_type(4)));
typedef int   i32x8   __attribute__((ext_vector_type(8)));

union AFrag { bf16x16 v; u32x4 q[2]; };

__device__ __forceinline__ unsigned short f32_to_bf16_rne(float f) {
    unsigned int x = __float_as_uint(f);
    unsigned int r = x + 0x7FFFu + ((x >> 16) & 1u);
    return (unsigned short)(r >> 16);
}
__device__ __forceinline__ float silu(float x) {
    return x / (1.0f + __expf(-x));
}

// ---------------------------------------------------------------------------
// Weight repack: W (KxN f32 row-major) -> bf16 fragments in the CDNA5 16-bit
// 16x32 fragment layout (ISA 7.12.2), tile order [tn][kt][lane][16].
// lane l: n = l&15 ; k_base = 32*kt + (l<16 ? 0 : 8)
// element e: k = k_base + (e<8 ? e : e+8)
// => a lane's fragment is two contiguous 16B chunks; an N-group of tiles is
//    one fully contiguous block (1-D TDM-copyable into LDS).
// ---------------------------------------------------------------------------
__global__ void pack_b_kernel(const float* __restrict__ W,
                              unsigned short* __restrict__ Wp,
                              int K, int N) {
    int idx = blockIdx.x * 256 + threadIdx.x;
    int total = K * N;
    if (idx >= total) return;
    int Kt   = K >> 5;
    int e    = idx & 15;
    int l    = (idx >> 4) & 31;
    int rest = idx >> 9;
    int kt   = rest % Kt;
    int tn   = rest / Kt;
    int n    = tn * 16 + (l & 15);
    int k    = kt * 32 + ((l < 16) ? 0 : 8) + ((e < 8) ? e : e + 8);
    Wp[idx] = f32_to_bf16_rne(W[(size_t)k * N + n]);
}

// ---------------------------------------------------------------------------
// Stage a contiguous packed-B block (elems bf16) into LDS.
// Preferred path: Tensor Data Mover (1-D descriptor), issued by wave 0 only
// (TDM ignores EXEC; one issue per block), completed with s_wait_tensorcnt.
// Fallback: cooperative uint4 copy.
// ---------------------------------------------------------------------------
__device__ __forceinline__ void stage_b_to_lds(unsigned short* lds_dst,
                                               const unsigned short* gsrc,
                                               int elems) {
#if defined(__has_builtin) && __has_builtin(__builtin_amdgcn_tensor_load_to_lds) && \
    __has_builtin(__builtin_amdgcn_s_wait_tensorcnt)
    if (threadIdx.x < 32) {
        unsigned long long ga = (unsigned long long)(uintptr_t)gsrc;
        unsigned int lds_off  = (unsigned int)(uintptr_t)lds_dst; // LDS aperture: low 32b = LDS byte addr
        // D# group0: count=1 (valid), lds_addr, global_addr[56:0], type=2
        u32x4 g0;
        g0[0] = 1u;
        g0[1] = lds_off;
        g0[2] = (unsigned int)(ga & 0xFFFFFFFFu);
        g0[3] = (unsigned int)((ga >> 32) & 0x01FFFFFFu) | (2u << 30);
        // D# group1: data_size=1 (2B); 1-D: tensor_dim0 = tile_dim0 = elems,
        // tensor_dim1 = 1, tile_dim1/2 = 0 (unused), dim0 stride = elems.
        i32x8 g1;
        g1[0] = 1 << 16;                                   // data_size=2B
        g1[1] = (int)((unsigned)(elems & 0xFFFF) << 16);   // tensor_dim0[15:0]
        g1[2] = (int)(((unsigned)elems >> 16) & 0xFFFFu)   // tensor_dim0[31:16]
              | (1 << 16);                                 // tensor_dim1 = 1
        g1[3] = (int)((unsigned)(elems & 0xFFFF) << 16);   // tile_dim0
        g1[4] = 0;                                         // tile_dim1/2 = 0
        g1[5] = elems;                                     // dim0_stride[31:0]
        g1[6] = 0;
        g1[7] = 0;
        i32x4 gz = {0, 0, 0, 0};
#if __clang_major__ >= 23
        i32x8 gz8 = {0, 0, 0, 0, 0, 0, 0, 0};
        __builtin_amdgcn_tensor_load_to_lds(g0, g1, gz, gz, gz8, 0);
#else
        __builtin_amdgcn_tensor_load_to_lds(g0, g1, gz, gz, 0);
#endif
        __builtin_amdgcn_s_wait_tensorcnt(0);
    }
#else
    const u32x4* s = reinterpret_cast<const u32x4*>(gsrc);
    u32x4*       d = reinterpret_cast<u32x4*>(lds_dst);
    int nvec = elems >> 3;                                 // 8 bf16 per uint4
    for (int i = threadIdx.x; i < nvec; i += 256) d[i] = s[i];
#endif
    __syncthreads();
}

// ---------------------------------------------------------------------------
// WMMA GEMM: C[M,N] = act( A(bf16) * Bp(packed bf16) [+ bias] [+ Cin] )
// Block = 8 waves = 8 M-tiles sharing one NT*16-wide N-group whose packed-B
// block is staged once into LDS (TDM). Each wave computes 16 x NT*16: one A
// fragment load feeds NT v_wmma_f32_16x16x32_bf16 per K-step; B fragments
// come from LDS via ds_load_b128. NT is compile-time so every WMMA is
// unconditional -> accumulators stay in fixed VGPR blocks (no phis/spills).
// ---------------------------------------------------------------------------
template<int NT>
__global__ void gemm_bf16_wmma(const unsigned short* __restrict__ A,
                               const unsigned short* __restrict__ Bp,
                               const float* __restrict__ Cin,
                               const float* __restrict__ bias,
                               float* __restrict__ C,
                               int N, int K, int addC, int act,
                               int ng0, int ngrid) {
    extern __shared__ unsigned short Bsh[];
    int Kt = K >> 5;
    int mg = blockIdx.x / ngrid;             // M-group of 8 tiles
    int ng = ng0 + blockIdx.x % ngrid;       // N-group (tiles ng*4 .. +NT-1)
    int wid  = threadIdx.x >> 5;
    int lane = threadIdx.x & 31;

    // Stage this N-group's packed B (contiguous) into LDS.
    stage_b_to_lds(Bsh, Bp + (size_t)(ng * 4) * Kt * 512, NT * Kt * 512);

    int tm      = mg * 8 + wid;              // M == NTOK, exact multiple of 128
    int m       = lane & 15;
    int koff    = (lane < 16) ? 0 : 8;
    size_t arow = (size_t)(tm * 16 + m) * K;

    f32x8 acc[NT];
    #pragma unroll
    for (int j = 0; j < NT; ++j) acc[j] = f32x8{0.f,0.f,0.f,0.f,0.f,0.f,0.f,0.f};

    for (int kt = 0; kt < Kt; ++kt) {
        const unsigned short* ap = A + arow + kt * 32 + koff;
        AFrag a;
        a.q[0] = *reinterpret_cast<const u32x4*>(ap);        // k = base..base+7
        a.q[1] = *reinterpret_cast<const u32x4*>(ap + 16);   // k = base+16..+23
        if (kt + 1 < Kt)
            __builtin_prefetch(ap + 32, 0, 1);               // global_prefetch
        const unsigned short* brow = Bsh + ((size_t)kt * 32 + lane) * 16;
        #pragma unroll
        for (int j = 0; j < NT; ++j) {
            const unsigned short* bp = brow + (size_t)j * Kt * 512;
            AFrag b;
            b.q[0] = *reinterpret_cast<const u32x4*>(bp);
            b.q[1] = *reinterpret_cast<const u32x4*>(bp + 8);
            acc[j] = __builtin_amdgcn_wmma_f32_16x16x32_bf16(
                         false, a.v, false, b.v, (short)0, acc[j],
                         false, false);
        }
    }

    // C/D layout: lane -> (n = lane&15, m = r + (lane<16?0:8))
    int n     = lane & 15;
    int mbase = (lane < 16) ? 0 : 8;
    #pragma unroll
    for (int j = 0; j < NT; ++j) {
        int col  = (ng * 4 + j) * 16 + n;
        float bv = bias ? bias[col] : 0.0f;
        #pragma unroll
        for (int r = 0; r < 8; ++r) {
            size_t off = (size_t)(tm * 16 + mbase + r) * N + col;
            float v = acc[j][r] + bv;
            if (addC) v += Cin[off];
            if (act == 1) v = silu(v);
            C[off] = v;
        }
    }
}

// ---------------------------------------------------------------------------
// Token prep: quat->rotmat, inverse-rotate force, build 10-feat vector,
// project to D_MODEL. One block per token; lane0 builds R/features in LDS.
// ---------------------------------------------------------------------------
__global__ void prep_kernel(const float* __restrict__ f3,
                            const float* __restrict__ q4,
                            const float* __restrict__ v3,
                            const float* __restrict__ projW,   // 10 x 256
                            const float* __restrict__ projB,   // 256
                            float* __restrict__ X,             // NTOK x 256
                            float* __restrict__ Rout,          // NTOK x 9
                            int writeR) {
    int tok = blockIdx.x;
    __shared__ float feat[10];
    __shared__ float R[9];
    if (threadIdx.x == 0) {
        float qw = q4[tok*4+0], qx = q4[tok*4+1], qy = q4[tok*4+2], qz = q4[tok*4+3];
        float inv = rsqrtf(qw*qw + qx*qx + qy*qy + qz*qz + 1e-30f);
        float w = qw*inv, x = qx*inv, y = qy*inv, z = qz*inv;
        R[0] = 1.f - 2.f*(y*y + z*z); R[1] = 2.f*(x*y - w*z); R[2] = 2.f*(x*z + w*y);
        R[3] = 2.f*(x*y + w*z); R[4] = 1.f - 2.f*(x*x + z*z); R[5] = 2.f*(y*z - w*x);
        R[6] = 2.f*(x*z - w*y); R[7] = 2.f*(y*z + w*x); R[8] = 1.f - 2.f*(x*x + y*y);
        float f0 = f3[tok*3+0], f1 = f3[tok*3+1], f2 = f3[tok*3+2];
        // einsum('ji,j->i'): rot_i = sum_j R[j][i] * f[j]
        feat[0] = R[0]*f0 + R[3]*f1 + R[6]*f2;
        feat[1] = R[1]*f0 + R[4]*f1 + R[7]*f2;
        feat[2] = R[2]*f0 + R[5]*f1 + R[8]*f2;
        feat[3] = qw; feat[4] = qx; feat[5] = qy; feat[6] = qz;   // raw q
        feat[7] = v3[tok*3+0]; feat[8] = v3[tok*3+1]; feat[9] = v3[tok*3+2];
    }
    __syncthreads();
    if (writeR && threadIdx.x < 9)
        Rout[(size_t)tok*9 + threadIdx.x] = R[threadIdx.x];
    int d = threadIdx.x;                       // blockDim == 256
    float s = projB[d];
    #pragma unroll
    for (int i = 0; i < 10; ++i) s += feat[i] * projW[i*D_MODEL + d];
    X[(size_t)tok*D_MODEL + d] = s;
}

// ---------------------------------------------------------------------------
// RMSNorm (row = token, 256 threads) -> bf16 output for WMMA.
// ---------------------------------------------------------------------------
__global__ void rmsnorm_kernel(const float* __restrict__ X,
                               const float* __restrict__ w,
                               unsigned short* __restrict__ XN) {
    int tok = blockIdx.x;
    int d   = threadIdx.x;
    float x = X[(size_t)tok*D_MODEL + d];
    __shared__ float red[256];
    red[d] = x * x;
    __syncthreads();
    for (int s = 128; s > 0; s >>= 1) {
        if (d < s) red[d] += red[d + s];
        __syncthreads();
    }
    float scale = rsqrtf(red[0] * (1.0f/D_MODEL) + 1e-5f);
    XN[(size_t)tok*D_MODEL + d] = f32_to_bf16_rne(x * scale * w[d]);
}

// ---------------------------------------------------------------------------
// Depthwise causal conv (k=4) + SiLU over xp = XZ[:, :512]; emits f32 + bf16.
// ---------------------------------------------------------------------------
__global__ void conv_silu_kernel(const float* __restrict__ XZ,   // NTOK x 1024
                                 const float* __restrict__ cw,   // 512 x 4
                                 const float* __restrict__ cb,   // 512
                                 float* __restrict__ XP,
                                 unsigned short* __restrict__ XPh) {
    size_t i = (size_t)blockIdx.x * 256 + threadIdx.x;           // NTOK*512
    int    d   = (int)(i & 511);
    size_t tok = i >> 9;
    int    t   = (int)(tok & (SEQ - 1));
    float acc = cb[d];
    #pragma unroll
    for (int j = 0; j < D_CONV; ++j) {
        int tt = t - (D_CONV - 1) + j;
        if (tt >= 0)
            acc += XZ[(tok - (D_CONV - 1) + j) * (2*D_INNER) + d] * cw[d*D_CONV + j];
    }
    float s = silu(acc);
    XP[i]  = s;
    XPh[i] = f32_to_bf16_rne(s);
}

// ---------------------------------------------------------------------------
// dt = softplus(dt_raw @ W_dt + b_dt), dt_raw = DBC[:, :16]  (tiny K: VALU)
// ---------------------------------------------------------------------------
__global__ void dt_kernel(const float* __restrict__ DBC,     // NTOK x 48
                          const float* __restrict__ Wdt,     // 16 x 512
                          const float* __restrict__ bdt,     // 512
                          float* __restrict__ DT) {
    size_t i = (size_t)blockIdx.x * 256 + threadIdx.x;       // NTOK*512
    int    d   = (int)(i & 511);
    size_t tok = i >> 9;
    const float* dbc = DBC + tok * DBC_DIM;
    float s = bdt[d];
    #pragma unroll
    for (int r = 0; r < DT_RANK; ++r) s += dbc[r] * Wdt[r*D_INNER + d];
    DT[i] = (s > 20.0f) ? s : logf(1.0f + __expf(s));
}

// ---------------------------------------------------------------------------
// Selective scan: one thread per (batch, channel); 16-state in registers,
// sequential over L=2048. B_t/C_t broadcast from DBC (L2-resident).
// ---------------------------------------------------------------------------
__global__ void scan_kernel(const float* __restrict__ DT,
                            const float* __restrict__ DBC,
                            const float* __restrict__ XP,
                            const float* __restrict__ A_log,  // 512 x 16
                            float* __restrict__ YS) {
    int tid = blockIdx.x * blockDim.x + threadIdx.x;          // 0..4095
    int b = tid >> 9;
    int d = tid & 511;
    float A[D_STATE], h[D_STATE];
    #pragma unroll
    for (int n = 0; n < D_STATE; ++n) {
        A[n] = -__expf(A_log[d*D_STATE + n]);
        h[n] = 0.0f;
    }
    for (int t = 0; t < SEQ; ++t) {
        size_t tok = (size_t)b * SEQ + t;
        float dt  = DT[tok*D_INNER + d];
        float x   = XP[tok*D_INNER + d];
        float dtx = dt * x;
        const float* bc = DBC + tok * DBC_DIM;
        float y = 0.0f;
        #pragma unroll
        for (int n = 0; n < D_STATE; ++n) {
            float dA = __expf(dt * A[n]);
            h[n] = dA * h[n] + dtx * bc[DT_RANK + n];
            y   += h[n] * bc[DT_RANK + D_STATE + n];
        }
        YS[tok*D_INNER + d] = y;
    }
}

// ---------------------------------------------------------------------------
// y = (ys + xp*D) * silu(z)  -> bf16 for the W_out WMMA GEMM
// ---------------------------------------------------------------------------
__global__ void gate_kernel(const float* __restrict__ YS,
                            const float* __restrict__ XP,
                            const float* __restrict__ XZ,     // z = cols 512..1023
                            const float* __restrict__ Dp,     // 512
                            unsigned short* __restrict__ Yh) {
    size_t i = (size_t)blockIdx.x * 256 + threadIdx.x;        // NTOK*512
    int    d   = (int)(i & 511);
    size_t tok = i >> 9;
    float z = XZ[tok*(2*D_INNER) + D_INNER + d];
    float y = (YS[i] + XP[i] * Dp[d]) * silu(z);
    Yh[i] = f32_to_bf16_rne(y);
}

// ---------------------------------------------------------------------------
// Head second layer: out = H1(128) @ w2(128x3) + b2   (tiny N: VALU)
// ---------------------------------------------------------------------------
__global__ void head2_kernel(const float* __restrict__ H1,
                             const float* __restrict__ w2,    // 128 x 3
                             const float* __restrict__ b2,    // 3
                             float* __restrict__ out) {
    int tok = blockIdx.x * blockDim.x + threadIdx.x;
    if (tok >= NTOK) return;
    const float* h = H1 + (size_t)tok * MLP_HID;
    float s0 = b2[0], s1 = b2[1], s2 = b2[2];
    #pragma unroll 8
    for (int k = 0; k < MLP_HID; ++k) {
        float hv = h[k];
        s0 += hv * w2[k*3 + 0];
        s1 += hv * w2[k*3 + 1];
        s2 += hv * w2[k*3 + 2];
    }
    out[(size_t)tok*3 + 0] = s0;
    out[(size_t)tok*3 + 1] = s1;
    out[(size_t)tok*3 + 2] = s2;
}

// ---------------------------------------------------------------------------
// Host orchestration
// ---------------------------------------------------------------------------
struct LayerW {
    const float *A_log, *D, *W_dt, *W_in, *W_out, *W_x, *b_dt, *conv_b, *conv_w, *norm_w;
};

extern "C" void kernel_launch(void* const* d_in, const int* in_sizes, int n_in,
                              void* d_out, int out_size, void* d_ws, size_t ws_size,
                              hipStream_t stream) {
    (void)in_sizes; (void)n_in; (void)out_size; (void)ws_size;
    auto F = [&](int i) { return (const float*)d_in[i]; };

    // Flattening: fw fb vw vb q | params pytree (dict keys alphabetical):
    // body_head{b1,b2,w1,w2}=5..8, body_layers[4]x10=9..48 (each layer:
    // A_log,D,W_dt,W_in,W_out,W_x,b_dt,conv_b,conv_w,norm_w), body_norm=49,
    // body_proj_b=50, body_proj_w=51, world_head=52..55, world_layers=56..95,
    // world_norm=96, world_proj_b=97, world_proj_w=98.
    const float* fw = F(0); const float* fb = F(1);
    const float* vw = F(2); const float* vb = F(3);
    const float* q  = F(4);

    auto layer_at = [&](int base) {
        LayerW L;
        L.A_log  = F(base+0); L.D      = F(base+1); L.W_dt = F(base+2);
        L.W_in   = F(base+3); L.W_out  = F(base+4); L.W_x  = F(base+5);
        L.b_dt   = F(base+6); L.conv_b = F(base+7); L.conv_w = F(base+8);
        L.norm_w = F(base+9);
        return L;
    };

    // Workspace carve-up (256B aligned)
    char* wsp = (char*)d_ws;
    auto alloc = [&](size_t bytes) {
        void* p = wsp; wsp += (bytes + 255) & ~(size_t)255; return p;
    };
    float*          X   = (float*)         alloc((size_t)NTOK * D_MODEL * 4);
    unsigned short* XN  = (unsigned short*)alloc((size_t)NTOK * D_MODEL * 2);
    float*          XZ  = (float*)         alloc((size_t)NTOK * 2*D_INNER * 4);
    float*          XP  = (float*)         alloc((size_t)NTOK * D_INNER * 4);
    unsigned short* XPh = (unsigned short*)alloc((size_t)NTOK * D_INNER * 2);
    float*          DBC = (float*)         alloc((size_t)NTOK * DBC_DIM * 4);
    float*          DT  = (float*)         alloc((size_t)NTOK * D_INNER * 4);
    float*          YS  = (float*)         alloc((size_t)NTOK * D_INNER * 4);
    unsigned short* Yh  = (unsigned short*)alloc((size_t)NTOK * D_INNER * 2);
    float*          H1  = (float*)         alloc((size_t)NTOK * MLP_HID * 4);
    unsigned short* Wp  = (unsigned short*)alloc((size_t)2 * 1024 * 1024);

    float* out   = (float*)d_out;
    float* out_R = out + 2 * NTOK * 3;                 // R after both F_hats

    auto run_gemm = [&](const unsigned short* A, const float* W, int K, int N,
                        float* C, const float* Cin, int addC,
                        const float* bias, int act) {
        int total = K * N;
        pack_b_kernel<<<(total + 255) / 256, 256, 0, stream>>>(W, Wp, K, N);
        int tiles_n = N / 16;
        int full    = tiles_n / 4;                     // full 4-tile N-groups
        int rem     = tiles_n % 4;
        int mgroups = (NTOK / 16) / 8;                 // 8 waves (wave32)/block
        int Kt      = K / 32;
        if (full > 0)
            gemm_bf16_wmma<4><<<mgroups * full, 256,
                                (size_t)4 * Kt * 512 * 2, stream>>>(
                A, Wp, Cin, bias, C, N, K, addC, act, 0, full);
        if (rem == 3)
            gemm_bf16_wmma<3><<<mgroups, 256, (size_t)3 * Kt * 512 * 2, stream>>>(
                A, Wp, Cin, bias, C, N, K, addC, act, full, 1);
        else if (rem == 2)
            gemm_bf16_wmma<2><<<mgroups, 256, (size_t)2 * Kt * 512 * 2, stream>>>(
                A, Wp, Cin, bias, C, N, K, addC, act, full, 1);
        else if (rem == 1)
            gemm_bf16_wmma<1><<<mgroups, 256, (size_t)1 * Kt * 512 * 2, stream>>>(
                A, Wp, Cin, bias, C, N, K, addC, act, full, 1);
    };

    const int ew_blocks = (NTOK * D_INNER) / 256;      // elementwise over 512ch

    for (int enc = 0; enc < 2; ++enc) {
        const bool world = (enc == 0);
        const float* f3    = world ? fw    : fb;
        const float* v3    = world ? vw    : vb;
        const float* projW = world ? F(98) : F(51);
        const float* projB = world ? F(97) : F(50);
        const float* encNw = world ? F(96) : F(49);
        const float* h_b1  = world ? F(52) : F(5);
        const float* h_b2  = world ? F(53) : F(6);
        const float* h_w1  = world ? F(54) : F(7);
        const float* h_w2  = world ? F(55) : F(8);
        const int layer0   = world ? 56 : 9;
        float* Fout        = out + (world ? 0 : NTOK * 3);

        prep_kernel<<<NTOK, 256, 0, stream>>>(f3, q, v3, projW, projB,
                                              X, out_R, world ? 1 : 0);

        for (int l = 0; l < N_LAYERS; ++l) {
            LayerW Lw = layer_at(layer0 + l * 10);
            rmsnorm_kernel<<<NTOK, 256, 0, stream>>>(X, Lw.norm_w, XN);
            run_gemm(XN, Lw.W_in, D_MODEL, 2*D_INNER, XZ, nullptr, 0, nullptr, 0);
            conv_silu_kernel<<<ew_blocks, 256, 0, stream>>>(XZ, Lw.conv_w,
                                                            Lw.conv_b, XP, XPh);
            run_gemm(XPh, Lw.W_x, D_INNER, DBC_DIM, DBC, nullptr, 0, nullptr, 0);
            dt_kernel<<<ew_blocks, 256, 0, stream>>>(DBC, Lw.W_dt, Lw.b_dt, DT);
            scan_kernel<<<(BATCH * D_INNER) / 256, 256, 0, stream>>>(
                DT, DBC, XP, Lw.A_log, YS);
            gate_kernel<<<ew_blocks, 256, 0, stream>>>(YS, XP, XZ, Lw.D, Yh);
            // residual add fused in GEMM epilogue: X = X + Yh @ W_out
            run_gemm(Yh, Lw.W_out, D_INNER, D_MODEL, X, X, 1, nullptr, 0);
        }

        rmsnorm_kernel<<<NTOK, 256, 0, stream>>>(X, encNw, XN);
        run_gemm(XN, h_w1, D_MODEL, MLP_HID, H1, nullptr, 0, h_b1, /*silu*/1);
        head2_kernel<<<(NTOK + 255) / 256, 256, 0, stream>>>(H1, h_w2, h_b2, Fout);
    }
}
